// MotionTransformerWithMultiEgoAggregation_67860483277512
// MI455X (gfx1250) — matrix operand order repaired and checked by reading, not compiled, not run on hardware
//
#include <hip/hip_runtime.h>

// ---------------------------------------------------------------------------
// Types for CDNA5 WMMA (wave32): v_wmma_f32_16x16x32_bf16
// ---------------------------------------------------------------------------
typedef __bf16  v16bf __attribute__((ext_vector_type(16)));
typedef __bf16  v8bf  __attribute__((ext_vector_type(8)));
typedef float   v8f   __attribute__((ext_vector_type(8)));

__device__ __forceinline__ unsigned bfround(float f) {
    unsigned u = __builtin_bit_cast(unsigned, f);
    return (u + 0x7FFFu + ((u >> 16) & 1u)) >> 16;   // round-to-nearest-even
}
__device__ __forceinline__ unsigned pk2(float lo, float hi) {  // [k] low, [k+1] high
    return bfround(lo) | (bfround(hi) << 16);
}
__device__ __forceinline__ __bf16 f2bf(float f) {
    unsigned short s = (unsigned short)bfround(f);
    return __builtin_bit_cast(__bf16, s);
}
__device__ __forceinline__ v16bf cat16(v8bf lo, v8bf hi) {
    return __builtin_shufflevector(lo, hi, 0,1,2,3,4,5,6,7,8,9,10,11,12,13,14,15);
}

// ---------------------------------------------------------------------------
// Problem constants
// ---------------------------------------------------------------------------
#define NC 4
#define NO 16
#define NM 6
#define NT 50
#define NS 5
#define DD 512
#define NNODES (NC*NO*NM)      // 384
#define MAPW 12544             // 49*256
#define FUSEDW 4224            // 6*512 + 128 + 1024
#define HH 48
#define WW 176
#define HW 8448                // 48*176
#define CIN 256
#define COUT 128
#define PADW 178               // WW+2
#define PADROWS (50*178)       // 8900

// ---------------------------------------------------------------------------
// Workspace layout (bytes)
// ---------------------------------------------------------------------------
static constexpr size_t OFF_A0    = 0;                                  // bf16 [384,256]
static constexpr size_t OFF_FEH1  = OFF_A0    + 384*256*2;
static constexpr size_t OFF_FEH2  = OFF_FEH1  + 384*256*2;
static constexpr size_t OFF_EMB   = OFF_FEH2  + 384*256*2;              // f32 [384,512]
static constexpr size_t OFF_MEAN  = OFF_EMB   + (size_t)384*512*4;      // f32 [512]
static constexpr size_t OFF_OUTV  = OFF_MEAN  + 512*4;                  // f32 [512]
static constexpr size_t OFF_MAPF  = OFF_OUTV  + 512*4;                  // bf16 [16,12544]
static constexpr size_t OFF_MEH1  = OFF_MAPF  + (size_t)16*MAPW*2;
static constexpr size_t OFF_MEH2  = OFF_MEH1  + (size_t)16*MAPW*2;
static constexpr size_t OFF_XPAD  = OFF_MEH2  + (size_t)16*MAPW*2;      // bf16 [8900,256]
static constexpr size_t OFF_WT    = OFF_XPAD  + (size_t)PADROWS*CIN*2;  // bf16 [9,256,128]
static constexpr size_t OFF_CONV  = OFF_WT    + (size_t)9*CIN*COUT*2;   // f32 [8448,128]
static constexpr size_t OFF_Y     = OFF_CONV  + (size_t)HW*COUT*4;      // f32 [128]
static constexpr size_t OFF_FUSED = OFF_Y     + 128*4;                  // bf16 [16,4224]
static constexpr size_t OFF_TDH1  = OFF_FUSED + (size_t)16*FUSEDW*2;
static constexpr size_t OFF_TDH2  = OFF_TDH1  + (size_t)16*FUSEDW*2;
static constexpr size_t OFF_CEH1  = OFF_TDH2  + (size_t)16*FUSEDW*2;
static constexpr size_t OFF_CEH2  = OFF_CEH1  + (size_t)16*FUSEDW*2;
static constexpr size_t OFF_LOGIT = OFF_CEH2  + (size_t)16*FUSEDW*2;    // f32 [16,6]
static constexpr size_t OFF_CACC  = (OFF_LOGIT + 96*4 + 255) & ~size_t(255); // f32 [16,12544] max

// ---------------------------------------------------------------------------
// Generic bf16-WMMA GEMM with optional split-K.
// 4-wave workgroup -> 16(M) x 64(N) tile; blockIdx.z = K-chunk.
// Double-buffered LDS; A row-major, B transposed in LDS so every lane's
// fragment is 2x contiguous ds_load_b128 (ISA 7.12.2 layouts).
// accmode=1: global_atomic_add_f32 partials into Cacc[M,naccw]; bias/act in
// a separate finish pass.  accmode=0 (ksplit==1): direct epilogue.
// ---------------------------------------------------------------------------
__global__ void k_gemm_bf16(const __bf16* __restrict__ A, int lda,
                            const float* __restrict__ B, int ldb, int K,
                            const float* __restrict__ bias,
                            void* __restrict__ Cout, int ldc, int coff,
                            int N, int Nfill, int Kpad, int relu, int outbf,
                            float* __restrict__ Cacc, int naccw,
                            int ktchunk, int accmode)
{
    __shared__ __bf16 sA [2][16][32];     // [m][k]
    __shared__ __bf16 sBt[2][64][32];     // [n][k]  (transposed)
    const int tid  = threadIdx.x;
    const int wave = tid >> 5;
    const int lane = tid & 31;
    const int half = lane >> 4;
    const int l16  = lane & 15;
    const int m0   = blockIdx.y * 16;
    const int n0   = blockIdx.x * 64;
    const int ar = tid >> 3, ac = (tid & 7) * 4;        // A staging: b64/thread
    const int kp = tid >> 3, cg = (tid & 7) * 8;        // B staging: k-pair, 8 cols

    const int ktiles = Kpad >> 5;
    const int kt0 = blockIdx.z * ktchunk;
    const int kt1 = (kt0 + ktchunk < ktiles) ? kt0 + ktchunk : ktiles;
    if (kt0 >= kt1) return;

    uint2    areg;
    unsigned bpack[8];

    auto stage_load = [&](int t) {
        const int k0  = t << 5;
        areg = *(const uint2*)(A + (size_t)(m0 + ar) * lda + k0 + ac);
        const int kr0 = k0 + 2 * kp, kr1 = kr0 + 1;
        if ((n0 + 64 <= ldb) & (k0 + 32 <= K)) {        // uniform fast path
            const float* b0 = B + (size_t)kr0 * ldb + n0 + cg;
            const float* b1 = b0 + ldb;
            if (k0 + 64 <= K)
                __builtin_prefetch((const void*)(b0 + (size_t)32 * ldb), 0, 0);
            float4 x0 = *(const float4*)(b0);
            float4 x1 = *(const float4*)(b0 + 4);
            float4 y0 = *(const float4*)(b1);
            float4 y1 = *(const float4*)(b1 + 4);
            bpack[0] = pk2(x0.x, y0.x);  bpack[1] = pk2(x0.y, y0.y);
            bpack[2] = pk2(x0.z, y0.z);  bpack[3] = pk2(x0.w, y0.w);
            bpack[4] = pk2(x1.x, y1.x);  bpack[5] = pk2(x1.y, y1.y);
            bpack[6] = pk2(x1.z, y1.z);  bpack[7] = pk2(x1.w, y1.w);
        } else {                                        // edge tiles only
            #pragma unroll
            for (int c = 0; c < 8; ++c) {
                int col = n0 + cg + c;
                float f0 = (kr0 < K && col < ldb) ? B[(size_t)kr0 * ldb + col] : 0.0f;
                float f1 = (kr1 < K && col < ldb) ? B[(size_t)kr1 * ldb + col] : 0.0f;
                bpack[c] = pk2(f0, f1);
            }
        }
    };
    auto stage_store = [&](int t) {
        const int buf = t & 1;
        *(uint2*)(&sA[buf][ar][ac]) = areg;
        #pragma unroll
        for (int c = 0; c < 8; ++c)
            *(unsigned*)(&sBt[buf][cg + c][2 * kp]) = bpack[c];
    };

    v8f acc = {};
    const int ncol = wave * 16 + l16;
    auto compute = [&](int t) {
        const int buf = t & 1;
        v8bf alo = *(const v8bf*)(&sA [buf][l16 ][ 8 * half]);
        v8bf ahi = *(const v8bf*)(&sA [buf][l16 ][16 + 8 * half]);
        v8bf blo = *(const v8bf*)(&sBt[buf][ncol][ 8 * half]);
        v8bf bhi = *(const v8bf*)(&sBt[buf][ncol][16 + 8 * half]);
        acc = __builtin_amdgcn_wmma_f32_16x16x32_bf16(
                  false, cat16(alo, ahi), false, cat16(blo, bhi),
                  (short)0, acc, false, false);
    };

    stage_load(kt0);
    stage_store(kt0);
    __syncthreads();
    for (int t = kt0; t < kt1 - 1; ++t) {   // branch-free steady state
        stage_load(t + 1);
        compute(t);
        stage_store(t + 1);
        __syncthreads();
    }
    compute(kt1 - 1);

    const int n = n0 + ncol;
    if (accmode) {
        #pragma unroll
        for (int r = 0; r < 8; ++r) {
            int m = m0 + r + 8 * half;
            atomicAdd(&Cacc[(size_t)m * naccw + n], acc[r]);
        }
    } else if (n < Nfill) {
        float bv = (n < N) ? bias[n] : 0.0f;
        #pragma unroll
        for (int r = 0; r < 8; ++r) {                   // C/D: VGPR r -> M = r + 8*half
            int m = m0 + r + 8 * half;
            float v = (n < N) ? (acc[r] + bv) : 0.0f;
            if (relu && v < 0.0f) v = 0.0f;
            size_t idx = (size_t)m * ldc + coff + n;
            if (outbf) ((__bf16*)Cout)[idx] = f2bf(v);
            else       ((float*)Cout)[idx]  = v;
        }
    }
}

__global__ void k_zero(float* __restrict__ p, int n)
{
    int i = blockIdx.x * 256 + threadIdx.x;
    if (i < n) p[i] = 0.0f;
}

__global__ void k_finish(const float* __restrict__ Cacc, int naccw,
                         const float* __restrict__ bias,
                         void* __restrict__ Cout, int ldc, int coff,
                         int N, int Nfill, int relu, int outbf)
{
    int idx = blockIdx.x * 256 + threadIdx.x;
    int m = idx / Nfill, n = idx % Nfill;
    if (idx < gridDim.x * 256 && m < (1 << 30)) {}      // no-op, keep simple
    if (idx >= Nfill * 16 && Nfill * 16 <= idx) {}      // (M is always 16 in split-K path)
    if (m >= 16) return;
    float v = (n < N) ? (Cacc[(size_t)m * naccw + n] + bias[n]) : 0.0f;
    if (relu && v < 0.0f) v = 0.0f;
    size_t o = (size_t)m * ldc + coff + n;
    if (outbf) ((__bf16*)Cout)[o] = f2bf(v);
    else       ((float*)Cout)[o]  = v;
}

// ---------------------------------------------------------------------------
// 3x3 SAME conv as implicit GEMM: out[8448,128] = relu(sum_9 Xshift@Wk + b)
// grid (2, 528); 72 flattened K-tiles, same pipelined scheme.
// ---------------------------------------------------------------------------
__global__ void k_conv(const __bf16* __restrict__ xpad,
                       const __bf16* __restrict__ wt,
                       const float*  __restrict__ cb,
                       float* __restrict__ convout)
{
    __shared__ __bf16 sA [2][16][32];
    __shared__ __bf16 sBt[2][64][32];
    const int tid  = threadIdx.x;
    const int wave = tid >> 5;
    const int lane = tid & 31;
    const int half = lane >> 4;
    const int l16  = lane & 15;
    const int mt = blockIdx.y;
    const int h  = mt / 11;
    const int w0 = (mt % 11) * 16;
    const int n0 = blockIdx.x * 64;
    const int ar = tid >> 3, ac = (tid & 7) * 4;
    const int kp = tid >> 3, cg = (tid & 7) * 8;

    uint2    areg;
    unsigned bpack[8];

    auto stage_load = [&](int t) {
        const int off = t >> 3;
        const int k0  = (t & 7) << 5;
        const int baserow = (h + off / 3) * PADW + (off % 3) + w0;   // (h+1+dy), (1+dx)
        areg = *(const uint2*)(xpad + (size_t)(baserow + ar) * CIN + k0 + ac);
        const __bf16* p0 = wt + (size_t)off * CIN * COUT + (size_t)(k0 + 2 * kp) * COUT + n0 + cg;
        uint4 r0 = *(const uint4*)(p0);
        uint4 r1 = *(const uint4*)(p0 + COUT);
        const unsigned* u0 = (const unsigned*)&r0;
        const unsigned* u1 = (const unsigned*)&r1;
        #pragma unroll
        for (int c = 0; c < 8; ++c) {
            unsigned e0 = (u0[c >> 1] >> ((c & 1) * 16)) & 0xFFFFu;
            unsigned e1 = (u1[c >> 1] >> ((c & 1) * 16)) & 0xFFFFu;
            bpack[c] = e0 | (e1 << 16);
        }
    };
    auto stage_store = [&](int t) {
        const int buf = t & 1;
        *(uint2*)(&sA[buf][ar][ac]) = areg;
        #pragma unroll
        for (int c = 0; c < 8; ++c)
            *(unsigned*)(&sBt[buf][cg + c][2 * kp]) = bpack[c];
    };

    v8f acc = {};
    const int ncol = wave * 16 + l16;
    auto compute = [&](int t) {
        const int buf = t & 1;
        v8bf alo = *(const v8bf*)(&sA [buf][l16 ][ 8 * half]);
        v8bf ahi = *(const v8bf*)(&sA [buf][l16 ][16 + 8 * half]);
        v8bf blo = *(const v8bf*)(&sBt[buf][ncol][ 8 * half]);
        v8bf bhi = *(const v8bf*)(&sBt[buf][ncol][16 + 8 * half]);
        acc = __builtin_amdgcn_wmma_f32_16x16x32_bf16(
                  false, cat16(alo, ahi), false, cat16(blo, bhi),
                  (short)0, acc, false, false);
    };

    stage_load(0);
    stage_store(0);
    __syncthreads();
    for (int t = 0; t < 71; ++t) {
        stage_load(t + 1);
        compute(t);
        stage_store(t + 1);
        __syncthreads();
    }
    compute(71);

    int co = n0 + ncol;
    #pragma unroll
    for (int r = 0; r < 8; ++r) {
        int pix = h * WW + w0 + r + 8 * half;
        float v = acc[r] + cb[co];
        if (v < 0.0f) v = 0.0f;
        convout[(size_t)pix * COUT + co] = v;
    }
}

// ---------------------------------------------------------------------------
// Small prep / glue kernels
// ---------------------------------------------------------------------------
__global__ void k_prep_scaled(const float* __restrict__ feat,
                              const float* __restrict__ score,
                              __bf16* __restrict__ A0)
{
    int idx = blockIdx.x * 256 + threadIdx.x;     // [384,256]
    if (idx < 384 * 256) {
        int node = idx >> 8, col = idx & 255;
        float v = 0.0f;
        if (col < NT * NS) v = feat[(size_t)node * (NT*NS) + col] * score[node];
        A0[idx] = f2bf(v);
    }
}

__global__ void k_prep_map(const float* __restrict__ mpf,
                           const int*   __restrict__ mask,
                           __bf16* __restrict__ mf)
{
    int idx = blockIdx.x * 256 + threadIdx.x;     // [16,12544]
    if (idx < 16 * MAPW) {
        int o = idx / MAPW, j = idx % MAPW;
        int p = j >> 8, q = j & 255;
        float v = mpf[((size_t)o * 49 + p) * 256 + q];   // CAV 0 only
        v *= (float)mask[o * 49 + p];
        mf[idx] = f2bf(v);
    }
}

__global__ void k_prep_xpad(const float* __restrict__ bev, __bf16* __restrict__ xp)
{
    int idx = blockIdx.x * 256 + threadIdx.x;     // [8900,256] channel-last, padded
    if (idx < PADROWS * CIN) {
        int row = idx >> 8, ci = idx & 255;
        int h = row / PADW - 1, w = row % PADW - 1;
        float v = 0.0f;
        if (h >= 0 && h < HH && w >= 0 && w < WW)
            v = bev[(size_t)ci * HW + h * WW + w];
        xp[idx] = f2bf(v);
    }
}

__global__ void k_prep_wt(const float* __restrict__ cw, __bf16* __restrict__ wt)
{
    int idx = blockIdx.x * 256 + threadIdx.x;     // [9][256][128]
    if (idx < 9 * CIN * COUT) {
        int off = idx / (CIN*COUT);
        int r = idx % (CIN*COUT);
        int ci = r >> 7, co = r & 127;
        wt[idx] = f2bf(cw[(size_t)co * 2304 + ci * 9 + off]);
    }
}

// GCN on complete graph + self loops == broadcast of mean(X) @ W + b
__global__ void k_mean(const float* __restrict__ emb, float* __restrict__ meanv)
{
    int j = blockIdx.x * 256 + threadIdx.x;
    if (j < DD) {
        float s = 0.0f;
        for (int i = 0; i < NNODES; ++i) s += emb[(size_t)i * DD + j];
        meanv[j] = s * (1.0f / (float)NNODES);
    }
}

__global__ void k_gcnvec(const float* __restrict__ meanv,
                         const float* __restrict__ gw,
                         const float* __restrict__ gb,
                         float* __restrict__ outv)
{
    int j = blockIdx.x * 256 + threadIdx.x;
    if (j < DD) {
        float s = gb[j];
        for (int k = 0; k < DD; ++k) s += meanv[k] * gw[(size_t)k * DD + j];
        outv[j] = s;
    }
}

__global__ void k_yinit(const float* __restrict__ bb, float* __restrict__ y)
{
    int j = threadIdx.x;
    if (j < COUT) y[j] = bb[j];
}

// bev GEMV: y[128] += sum_k relu_conv[k] * bev_w[k,128]; streams 553 MB.
// 512 blocks x 128 thr; thread owns 4 cols (one float4/row), 4 k-rows/iter.
__global__ void k_bev_gemv(const float* __restrict__ convout,
                           const float* __restrict__ bw,
                           float* __restrict__ y)
{
    __shared__ float sred[4][128];
    const int tid = threadIdx.x;
    const int r   = tid >> 5;             // wave-uniform k-row in quad
    const int c4  = (tid & 31) * 4;       // 4 output columns
    const size_t kbase = (size_t)blockIdx.x * 2112;
    const int co  = (int)(kbase / HW);    // 8448 = 4*2112: chunk stays in one co
    const int hwb = (int)(kbase % HW);
    float a0 = 0.f, a1 = 0.f, a2 = 0.f, a3 = 0.f;
    for (int i = 0; i < 528; ++i) {
        int k = i * 4 + r;
        float xv = convout[(size_t)(hwb + k) * COUT + co];   // uniform -> scalar ld
        float4 w = *(const float4*)(bw + (kbase + k) * COUT + c4);
        a0 += xv * w.x; a1 += xv * w.y; a2 += xv * w.z; a3 += xv * w.w;
    }
    sred[r][c4+0] = a0; sred[r][c4+1] = a1; sred[r][c4+2] = a2; sred[r][c4+3] = a3;
    __syncthreads();
    float s = sred[0][tid] + sred[1][tid] + sred[2][tid] + sred[3][tid];
    atomicAdd(&y[tid], s);
}

// fused[:, 0:3072] = tile(gcn_vec, 6);  fused[:, 3072:3200] = bev_emb
__global__ void k_fused_head(const float* __restrict__ outv,
                             const float* __restrict__ y,
                             __bf16* __restrict__ fused)
{
    int idx = blockIdx.x * 256 + threadIdx.x;     // 16*3200
    if (idx < 16 * 3200) {
        int o = idx / 3200, j = idx % 3200;
        float v = (j < 3072) ? outv[j & 511] : y[j - 3072];
        fused[(size_t)o * FUSEDW + j] = f2bf(v);
    }
}

__global__ void k_softmax(const float* __restrict__ logits, float* __restrict__ out)
{
    int o = threadIdx.x;
    if (o < NO) {
        float mx = -1e30f;
        for (int m = 0; m < NM; ++m) mx = fmaxf(mx, logits[o*NM + m]);
        float e[NM]; float s = 0.0f;
        for (int m = 0; m < NM; ++m) { e[m] = __expf(logits[o*NM + m] - mx); s += e[m]; }
        float inv = 1.0f / s;
        for (int m = 0; m < NM; ++m) out[o*NM + m] = e[m] * inv;
    }
}

// ---------------------------------------------------------------------------
// Host side
// ---------------------------------------------------------------------------
static inline void gemm_layer(const __bf16* A, int lda, const float* B, int ldb,
                              int K, const float* bias, void* C, int ldc, int coff,
                              int N, int Nfill, int Kpad, int relu, int outbf,
                              int M, int ksplit, float* Cacc, hipStream_t s)
{
    const int ntiles = (Nfill + 63) / 64;
    const int mtiles = M / 16;
    const int ktiles = Kpad / 32;
    if (ksplit <= 1) {
        dim3 g(ntiles, mtiles, 1);
        k_gemm_bf16<<<g, 128, 0, s>>>(A, lda, B, ldb, K, bias, C, ldc, coff,
                                      N, Nfill, Kpad, relu, outbf,
                                      nullptr, 0, ktiles, 0);
    } else {
        const int naccw = ntiles * 64;
        const int zc = M * naccw;
        k_zero<<<(zc + 255) / 256, 256, 0, s>>>(Cacc, zc);
        const int ktchunk = (ktiles + ksplit - 1) / ksplit;
        dim3 g(ntiles, mtiles, ksplit);
        k_gemm_bf16<<<g, 128, 0, s>>>(A, lda, B, ldb, K, bias, C, ldc, coff,
                                      N, Nfill, Kpad, relu, outbf,
                                      Cacc, naccw, ktchunk, 1);
        const int fc = M * Nfill;
        k_finish<<<(fc + 255) / 256, 256, 0, s>>>(Cacc, naccw, bias, C, ldc, coff,
                                                  N, Nfill, relu, outbf);
    }
}

extern "C" void kernel_launch(void* const* d_in, const int* in_sizes, int n_in,
                              void* d_out, int out_size, void* d_ws, size_t ws_size,
                              hipStream_t stream)
{
    (void)in_sizes; (void)n_in; (void)out_size; (void)ws_size;
    const float* feat   = (const float*)d_in[0];   // [4,16,6,50,5]
    const float* score  = (const float*)d_in[1];   // [4,16,6]
    const float* bev    = (const float*)d_in[2];   // [1,256,48,176]
    const float* mpf    = (const float*)d_in[5];   // [4,16,49,256]
    const int*   mmask  = (const int*)d_in[6];     // [4,16,49]
    const float* fe_w0  = (const float*)d_in[8];   const float* fe_b0 = (const float*)d_in[9];
    const float* fe_w1  = (const float*)d_in[10];  const float* fe_b1 = (const float*)d_in[11];
    const float* fe_w2  = (const float*)d_in[12];  const float* fe_b2 = (const float*)d_in[13];
    const float* gcn_w  = (const float*)d_in[14];  const float* gcn_b = (const float*)d_in[15];
    const float* conv_w = (const float*)d_in[16];  const float* conv_b= (const float*)d_in[17];
    const float* bev_w  = (const float*)d_in[18];  const float* bev_b = (const float*)d_in[19];
    const float* me_w0  = (const float*)d_in[20];  const float* me_b0 = (const float*)d_in[21];
    const float* me_w1  = (const float*)d_in[22];  const float* me_b1 = (const float*)d_in[23];
    const float* me_w2  = (const float*)d_in[24];  const float* me_b2 = (const float*)d_in[25];
    const float* td_w0  = (const float*)d_in[26];  const float* td_b0 = (const float*)d_in[27];
    const float* td_w1  = (const float*)d_in[28];  const float* td_b1 = (const float*)d_in[29];
    const float* td_w2  = (const float*)d_in[30];  const float* td_b2 = (const float*)d_in[31];
    const float* ce_w0  = (const float*)d_in[32];  const float* ce_b0 = (const float*)d_in[33];
    const float* ce_w1  = (const float*)d_in[34];  const float* ce_b1 = (const float*)d_in[35];
    const float* ce_w2  = (const float*)d_in[36];  const float* ce_b2 = (const float*)d_in[37];

    char* ws = (char*)d_ws;
    __bf16* A0    = (__bf16*)(ws + OFF_A0);
    __bf16* feh1  = (__bf16*)(ws + OFF_FEH1);
    __bf16* feh2  = (__bf16*)(ws + OFF_FEH2);
    float*  emb   = (float*) (ws + OFF_EMB);
    float*  meanv = (float*) (ws + OFF_MEAN);
    float*  outv  = (float*) (ws + OFF_OUTV);
    __bf16* mapf  = (__bf16*)(ws + OFF_MAPF);
    __bf16* meh1  = (__bf16*)(ws + OFF_MEH1);
    __bf16* meh2  = (__bf16*)(ws + OFF_MEH2);
    __bf16* xpad  = (__bf16*)(ws + OFF_XPAD);
    __bf16* wt    = (__bf16*)(ws + OFF_WT);
    float*  convo = (float*) (ws + OFF_CONV);
    float*  yvec  = (float*) (ws + OFF_Y);
    __bf16* fused = (__bf16*)(ws + OFF_FUSED);
    __bf16* tdh1  = (__bf16*)(ws + OFF_TDH1);
    __bf16* tdh2  = (__bf16*)(ws + OFF_TDH2);
    __bf16* ceh1  = (__bf16*)(ws + OFF_CEH1);
    __bf16* ceh2  = (__bf16*)(ws + OFF_CEH2);
    float*  logit = (float*) (ws + OFF_LOGIT);
    float*  cacc  = (float*) (ws + OFF_CACC);
    float*  traj  = (float*)d_out;                 // [16,6,50,5] flat
    float*  conf  = (float*)d_out + 16*NM*NT*NS;   // [16,6]

    // ---- prep ----
    k_prep_scaled<<<(384*256 + 255)/256, 256, 0, stream>>>(feat, score, A0);
    k_prep_map  <<<(16*MAPW + 255)/256, 256, 0, stream>>>(mpf, mmask, mapf);
    k_prep_xpad <<<(PADROWS*CIN + 255)/256, 256, 0, stream>>>(bev, xpad);
    k_prep_wt   <<<(9*CIN*COUT + 255)/256, 256, 0, stream>>>(conv_w, wt);

    // ---- feature encoder MLP: 250->250->250->512 over 384 nodes (no split) ----
    gemm_layer(A0,   256, fe_w0, 250, 250, fe_b0, feh1, 256, 0, 250, 256, 256, 1, 1, 384, 1, nullptr, stream);
    gemm_layer(feh1, 256, fe_w1, 250, 250, fe_b1, feh2, 256, 0, 250, 256, 256, 1, 1, 384, 1, nullptr, stream);
    gemm_layer(feh2, 256, fe_w2, 512, 250, fe_b2, emb,  512, 0, 512, 512, 256, 0, 0, 384, 1, nullptr, stream);

    // ---- GCN (complete graph + self loops) == mean @ W + b, broadcast ----
    k_mean  <<<2, 256, 0, stream>>>(emb, meanv);
    k_gcnvec<<<2, 256, 0, stream>>>(meanv, gcn_w, gcn_b, outv);

    // ---- bev: conv3x3 (implicit GEMM) -> relu -> giant GEMV ----
    k_conv<<<dim3(2, 528), 128, 0, stream>>>(xpad, wt, conv_b, convo);
    k_yinit<<<1, 128, 0, stream>>>(bev_b, yvec);
    k_bev_gemv<<<512, 128, 0, stream>>>(convo, bev_w, yvec);

    // ---- fused head: ego_nodes | bev_emb ----
    k_fused_head<<<(16*3200 + 255)/256, 256, 0, stream>>>(outv, yvec, fused);

    // ---- map encoder MLP: 12544->12544->12544->1024 (split-K) ----
    gemm_layer(mapf, MAPW, me_w0, MAPW, MAPW, me_b0, meh1, MAPW, 0, MAPW, MAPW, MAPW, 1, 1, 16, 8,  cacc, stream);
    gemm_layer(meh1, MAPW, me_w1, MAPW, MAPW, me_b1, meh2, MAPW, 0, MAPW, MAPW, MAPW, 1, 1, 16, 8,  cacc, stream);
    gemm_layer(meh2, MAPW, me_w2, 1024, MAPW, me_b2, fused, FUSEDW, 3200, 1024, 1024, MAPW, 0, 1, 16, 16, cacc, stream);

    // ---- trajectory decoder: 4224->4224->4224->1500 -> d_out (split-K) ----
    gemm_layer(fused, FUSEDW, td_w0, FUSEDW, FUSEDW, td_b0, tdh1, FUSEDW, 0, FUSEDW, FUSEDW, FUSEDW, 1, 1, 16, 16, cacc, stream);
    gemm_layer(tdh1,  FUSEDW, td_w1, FUSEDW, FUSEDW, td_b1, tdh2, FUSEDW, 0, FUSEDW, FUSEDW, FUSEDW, 1, 1, 16, 16, cacc, stream);
    gemm_layer(tdh2,  FUSEDW, td_w2, 1500,   FUSEDW, td_b2, traj, 1500,   0, 1500,   1500,   FUSEDW, 0, 0, 16, 16, cacc, stream);

    // ---- confidence estimator: 4224->4224->4224->6 (split-K) ----
    gemm_layer(fused, FUSEDW, ce_w0, FUSEDW, FUSEDW, ce_b0, ceh1, FUSEDW, 0, FUSEDW, FUSEDW, FUSEDW, 1, 1, 16, 16, cacc, stream);
    gemm_layer(ceh1,  FUSEDW, ce_w1, FUSEDW, FUSEDW, ce_b1, ceh2, FUSEDW, 0, FUSEDW, FUSEDW, FUSEDW, 1, 1, 16, 16, cacc, stream);
    gemm_layer(ceh2,  FUSEDW, ce_w2, NM,     FUSEDW, ce_b2, logit, NM,    0, NM,     NM,     FUSEDW, 0, 0, 16, 64, cacc, stream);
    k_softmax<<<1, 32, 0, stream>>>(logit, conf);
}
